// CoAtNetTransformerBlock_44092134261303
// MI455X (gfx1250) — matrix operand order repaired and checked by reading, not compiled, not run on hardware
//
#include <hip/hip_runtime.h>
#include <hip/hip_bf16.h>

#define DEV __device__ __forceinline__

typedef __attribute__((ext_vector_type(16))) _Float16 v16h;
typedef __attribute__((ext_vector_type(8)))  _Float16 v8h;
typedef __attribute__((ext_vector_type(8)))  float    v8f;

static constexpr int Hdim = 28, Wdim = 28, Cdim = 512, HEADS = 16;
static constexpr int Ndim = Hdim * Wdim;                 // 784
static constexpr int FF   = 4 * Cdim;                    // 2048
static constexpr int DH   = Cdim / HEADS;                // 32
static constexpr int NUM_REL = (2*Hdim-1)*Wdim + (2*Wdim-1) + 1; // 1596
static constexpr int BATCH = 16;
static constexpr int Mtot  = BATCH * Ndim;               // 12544
static constexpr int NPAD  = 800;                        // 784 padded to 25*32

// ---------------------------------------------------------------------------
// WMMA helpers (CDNA5 wave32, V_WMMA_F32_16X16X32_F16)
// ---------------------------------------------------------------------------
DEV v8f wmma_f16(v16h a, v16h b, v8f c) {
  return __builtin_amdgcn_wmma_f32_16x16x32_f16(
      /*neg_a=*/false, a, /*neg_b=*/false, b,
      /*c_mod=*/(short)0, c, /*reuse_a=*/false, /*reuse_b=*/false);
}

// A-matrix fragment (16x32 f16), row-major source, ld in elements.
// ISA layout: lanes 0-15 row M=lane, K={0..7,16..23}; lanes 16-31 K={8..15,24..31}.
DEV v16h frag_a(const _Float16* base, int ld, int lane) {
  const _Float16* p = base + (size_t)(lane & 15) * ld + ((lane >> 4) << 3);
  v8h lo = *(const v8h*)p;        // K koff .. koff+7
  v8h hi = *(const v8h*)(p + 16); // K koff+16 .. koff+23
  v16h r;
#pragma unroll
  for (int i = 0; i < 8; ++i) { r[i] = lo[i]; r[i + 8] = hi[i]; }
  return r;
}

// B-matrix fragment (32x16 f16) from TRANSPOSED storage BT[n][k], ld in elems.
// ISA layout: lanes 0-15 col N=lane K={0..7,16..23}; lanes 16-31 col N=lane-16 K={8..15,24..31}.
DEV v16h frag_bT(const _Float16* baseT, int ld, int lane) {
  const _Float16* p = baseT + (size_t)(lane & 15) * ld + ((lane >> 4) << 3);
  v8h lo = *(const v8h*)p;
  v8h hi = *(const v8h*)(p + 16);
  v16h r;
#pragma unroll
  for (int i = 0; i < 8; ++i) { r[i] = lo[i]; r[i + 8] = hi[i]; }
  return r;
}

// A fragment gathered from f32 LDS with conversion to f16.
DEV v16h frag_a_lds(const float* base, int ld, int lane) {
  const float* p = base + (size_t)(lane & 15) * ld + ((lane >> 4) << 3);
  v16h r;
#pragma unroll
  for (int i = 0; i < 8; ++i) {
    r[i]     = (_Float16)p[i];
    r[i + 8] = (_Float16)p[16 + i];
  }
  return r;
}

// ---------------------------------------------------------------------------
// LayerNorm: x (b,C,H,W) f32 -> xn (b*N, C) f16
// ---------------------------------------------------------------------------
__global__ __launch_bounds__(256) void ln_kernel(
    const float* __restrict__ x, const float* __restrict__ w,
    const float* __restrict__ b, _Float16* __restrict__ xn) {
  const int row = blockIdx.x;             // 0..Mtot-1
  const int bi  = row / Ndim;
  const int n   = row - bi * Ndim;
  const int t   = threadIdx.x;
  const float* xr = x + ((size_t)bi * Cdim) * Ndim + n;  // channel stride = Ndim
  float v0 = xr[(size_t)t * Ndim];
  float v1 = xr[(size_t)(t + 256) * Ndim];
  __shared__ float rs[256], rq[256];
  rs[t] = v0 + v1;
  rq[t] = v0 * v0 + v1 * v1;
  __syncthreads();
  for (int o = 128; o > 0; o >>= 1) {
    if (t < o) { rs[t] += rs[t + o]; rq[t] += rq[t + o]; }
    __syncthreads();
  }
  const float mean = rs[0] * (1.0f / Cdim);
  const float var  = rq[0] * (1.0f / Cdim) - mean * mean;
  const float inv  = rsqrtf(var + 1e-5f);
  _Float16* o = xn + (size_t)row * Cdim;
  o[t]       = (_Float16)((v0 - mean) * inv * w[t] + b[t]);
  o[t + 256] = (_Float16)((v1 - mean) * inv * w[t + 256] + b[t + 256]);
}

// ---------------------------------------------------------------------------
// Weight transpose + f32->f16: W (K x Nc) -> WT (Nc x K)
// ---------------------------------------------------------------------------
__global__ __launch_bounds__(256) void wT_f16(
    const float* __restrict__ W, _Float16* __restrict__ WT, int K, int Nc) {
  size_t i = (size_t)blockIdx.x * 256 + threadIdx.x;
  if (i < (size_t)K * Nc) {
    int kk = (int)(i / Nc), n = (int)(i - (size_t)kk * Nc);
    WT[(size_t)n * K + kk] = (_Float16)W[i];
  }
}

// Zero the N-padding region of the transposed V buffer (b*h, DH, NPAD).
__global__ __launch_bounds__(256) void vpad_zero(_Float16* __restrict__ v) {
  int i = blockIdx.x * 256 + threadIdx.x;       // BATCH*HEADS*DH*16 total
  int rowi = i >> 4, j = i & 15;
  v[(size_t)rowi * NPAD + Ndim + j] = (_Float16)0.0f;
}

// ---------------------------------------------------------------------------
// Generic WMMA GEMM: D = A(f16, MxK) @ BT(f16, Nc x K)^T + bias, with epilogue.
// Block: 128 threads = 4 waves. Each wave owns a 32x64 tile: 2 A-frags x
// 4 B-frags -> 8 WMMAs per K-step (2x B-fragment reuse vs 16x64 tiling).
// EPI: 0=Q/K scatter  1=V transposed scatter  2=O-proj+residual
//      3=GELU->f16    4=FFN2+residual->NCHW f32 out
// ---------------------------------------------------------------------------
template <int EPI>
__global__ __launch_bounds__(128) void gemm16(
    const _Float16* __restrict__ A, const _Float16* __restrict__ BT,
    const float* __restrict__ bias, int M, int Nc, int K,
    _Float16* __restrict__ outH, float* __restrict__ outF,
    const float* __restrict__ resid) {
  const int lane    = threadIdx.x & 31;
  const int wave    = threadIdx.x >> 5;
  const int tile_m  = blockIdx.x * 32;
  const int colbase = blockIdx.y * 256 + wave * 64;
  if (tile_m >= M || colbase >= Nc) return;   // wave-uniform guard

  v8f acc[2][4] = {};
  const _Float16* A0 = A + (size_t)tile_m * K;
  const _Float16* A1 = A0 + (size_t)16 * K;
  const _Float16* myArow = A0 + (size_t)(lane & 15) * K;  // for prefetch
  for (int kk = 0; kk < K; kk += 32) {
    if (kk + 32 < K)
      __builtin_prefetch(myArow + kk + 32, 0, 3);  // global_prefetch_b8
    v16h a0 = frag_a(A0 + kk, K, lane);
    v16h a1 = frag_a(A1 + kk, K, lane);
#pragma unroll
    for (int s = 0; s < 4; ++s) {
      v16h bF = frag_bT(BT + (size_t)(colbase + s * 16) * K + kk, K, lane);
      acc[0][s] = wmma_f16(a0, bF, acc[0][s]);
      acc[1][s] = wmma_f16(a1, bF, acc[1][s]);
    }
  }

  const int rbase = (lane >> 4) * 8;
  const int nloc  = lane & 15;
#pragma unroll
  for (int t = 0; t < 2; ++t) {
#pragma unroll
    for (int s = 0; s < 4; ++s) {
      const int gn  = colbase + s * 16 + nloc;
      const float bi = bias[gn];
#pragma unroll
      for (int vv = 0; vv < 8; ++vv) {
        const int gm = tile_m + t * 16 + rbase + vv;
        float val = acc[t][s][vv] + bi;
        const int bidx = gm / Ndim;
        const int n    = gm - bidx * Ndim;
        if (EPI == 0) {                       // Q/K -> (b,h,n,dh) f16
          const int h = gn >> 5, dh = gn & 31;
          outH[(((size_t)bidx * HEADS + h) * Ndim + n) * DH + dh] = (_Float16)val;
        } else if (EPI == 1) {                // V -> (b,h,dh,n) padded f16
          const int h = gn >> 5, dh = gn & 31;
          outH[(((size_t)bidx * HEADS + h) * DH + dh) * NPAD + n] = (_Float16)val;
        } else if (EPI == 2) {                // x1 = xc + attn@Wo + bo
          val += resid[((size_t)bidx * Cdim + gn) * Ndim + n]; // xc from NCHW x
          outF[(size_t)gm * Cdim + gn] = val;
          outH[(size_t)gm * Cdim + gn] = (_Float16)val;
        } else if (EPI == 3) {                // exact GELU -> f16
          float g = 0.5f * val * (1.0f + erff(val * 0.70710678118654752f));
          outH[(size_t)gm * FF + gn] = (_Float16)g;
        } else {                              // EPI==4: y = x1 + h@W2 + b2 -> NCHW
          val += resid[(size_t)gm * Cdim + gn];
          outF[((size_t)bidx * Cdim + gn) * Ndim + n] = val;
        }
      }
    }
  }
}

// ---------------------------------------------------------------------------
// Attention: one wave per (b*h, 16-row tile). scores=QK^T+relbias, softmax,
// out = attn @ V. Scores staged in 51KB static LDS (f32), attn A-fragments
// converted f32->f16 on the fly from LDS.
// ---------------------------------------------------------------------------
__global__ __launch_bounds__(32) void attn_kernel(
    const _Float16* __restrict__ q, const _Float16* __restrict__ k,
    const _Float16* __restrict__ v, const float* __restrict__ rel_bias,
    const int* __restrict__ rel_idx, _Float16* __restrict__ attn_out) {
  __shared__ float sc[16 * NPAD];   // 51200 bytes
  const int lane = threadIdx.x;
  const int rt   = blockIdx.x;      // 0..48
  const int bh   = blockIdx.y;      // 0..255
  const int h    = bh & (HEADS - 1);
  const int bi   = bh >> 4;
  const int rbase = (lane >> 4) * 8;
  const int nloc  = lane & 15;

  // ---- scores = q @ k^T + bias ----
  const _Float16* qbase = q + ((size_t)bh * Ndim + rt * 16) * DH;
  v16h aF = frag_a(qbase, DH, lane);       // single K=32 chunk (DH=32)
  for (int ct = 0; ct < Ndim / 16; ++ct) {
    const _Float16* kbase = k + ((size_t)bh * Ndim + ct * 16) * DH;
    v16h bF = frag_bT(kbase, DH, lane);    // k rows are BT rows
    v8f acc = {};
    acc = wmma_f16(aF, bF, acc);
#pragma unroll
    for (int vv = 0; vv < 8; ++vv) {
      const int r  = rbase + vv;
      const int n0 = rt * 16 + r;
      const int cg = ct * 16 + nloc;
      const float bia = rel_bias[h * NUM_REL + rel_idx[n0 * Ndim + cg]];
      sc[r * NPAD + cg] = acc[vv] + bia;
    }
  }
  __syncthreads();

  // ---- softmax (all 32 lanes cooperate per row) ----
  for (int r = 0; r < 16; ++r) {
    float mx = -1e30f;
    for (int j = lane; j < Ndim; j += 32) mx = fmaxf(mx, sc[r * NPAD + j]);
    for (int o = 16; o > 0; o >>= 1) mx = fmaxf(mx, __shfl_xor(mx, o, 32));
    float sum = 0.0f;
    for (int j = lane; j < Ndim; j += 32) {
      float e = __expf(sc[r * NPAD + j] - mx);
      sc[r * NPAD + j] = e;
      sum += e;
    }
    for (int o = 16; o > 0; o >>= 1) sum += __shfl_xor(sum, o, 32);
    const float inv = 1.0f / sum;
    for (int j = lane; j < NPAD; j += 32)
      sc[r * NPAD + j] = (j < Ndim) ? sc[r * NPAD + j] * inv : 0.0f;
  }
  __syncthreads();

  // ---- out = attn @ V  (K = NPAD = 25*32), V stored transposed (dh, n) ----
  v8f o0 = {}, o1 = {};
  const _Float16* vbh = v + (size_t)bh * DH * NPAD;
  for (int kc = 0; kc < NPAD / 32; ++kc) {
    v16h aT = frag_a_lds(sc + kc * 32, NPAD, lane);
    v16h b0 = frag_bT(vbh + kc * 32, NPAD, lane);              // dh 0..15
    v16h b1 = frag_bT(vbh + (size_t)16 * NPAD + kc * 32, NPAD, lane); // 16..31
    o0 = wmma_f16(aT, b0, o0);
    o1 = wmma_f16(aT, b1, o1);
  }
#pragma unroll
  for (int vv = 0; vv < 8; ++vv) {
    const int n0 = rt * 16 + rbase + vv;
    const size_t o = ((size_t)bi * Ndim + n0) * Cdim + h * DH;
    attn_out[o + nloc]      = (_Float16)o0[vv];
    attn_out[o + 16 + nloc] = (_Float16)o1[vv];
  }
}

// ---------------------------------------------------------------------------
// Host launcher
// ---------------------------------------------------------------------------
extern "C" void kernel_launch(void* const* d_in, const int* in_sizes, int n_in,
                              void* d_out, int out_size, void* d_ws, size_t ws_size,
                              hipStream_t stream) {
  (void)in_sizes; (void)n_in; (void)out_size; (void)ws_size;
  const float* x     = (const float*)d_in[0];
  const float* ln_w  = (const float*)d_in[1];
  const float* ln_b  = (const float*)d_in[2];
  const float* Wq    = (const float*)d_in[3];
  const float* bq    = (const float*)d_in[4];
  const float* Wk    = (const float*)d_in[5];
  const float* bk    = (const float*)d_in[6];
  const float* Wv    = (const float*)d_in[7];
  const float* bv    = (const float*)d_in[8];
  const float* Wo    = (const float*)d_in[9];
  const float* bo    = (const float*)d_in[10];
  const float* relb  = (const float*)d_in[11];
  const float* W1    = (const float*)d_in[12];
  const float* b1    = (const float*)d_in[13];
  const float* W2    = (const float*)d_in[14];
  const float* b2    = (const float*)d_in[15];
  const int*   relix = (const int*)d_in[16];
  float* out = (float*)d_out;

  char* ws = (char*)d_ws;
  size_t off = 0;
  auto alloc = [&](size_t bytes) -> void* {
    void* p = ws + off;
    off += (bytes + 255) & ~(size_t)255;
    return p;
  };
  _Float16* xn  = (_Float16*)alloc((size_t)Mtot * Cdim * 2);
  _Float16* qb  = (_Float16*)alloc((size_t)BATCH * HEADS * Ndim * DH * 2);
  _Float16* kb  = (_Float16*)alloc((size_t)BATCH * HEADS * Ndim * DH * 2);
  _Float16* vb  = (_Float16*)alloc((size_t)BATCH * HEADS * DH * NPAD * 2);
  _Float16* ao  = (_Float16*)alloc((size_t)Mtot * Cdim * 2);
  float*    x1  = (float*)   alloc((size_t)Mtot * Cdim * 4);
  _Float16* x1h = (_Float16*)alloc((size_t)Mtot * Cdim * 2);
  _Float16* hid = (_Float16*)alloc((size_t)Mtot * FF * 2);
  _Float16* WqT = (_Float16*)alloc((size_t)Cdim * Cdim * 2);
  _Float16* WkT = (_Float16*)alloc((size_t)Cdim * Cdim * 2);
  _Float16* WvT = (_Float16*)alloc((size_t)Cdim * Cdim * 2);
  _Float16* WoT = (_Float16*)alloc((size_t)Cdim * Cdim * 2);
  _Float16* W1T = (_Float16*)alloc((size_t)Cdim * FF * 2);
  _Float16* W2T = (_Float16*)alloc((size_t)FF * Cdim * 2);

  // 1) LayerNorm + f16 conversion
  ln_kernel<<<Mtot, 256, 0, stream>>>(x, ln_w, ln_b, xn);

  // 2) weight transpose/convert
  wT_f16<<<(Cdim * Cdim + 255) / 256, 256, 0, stream>>>(Wq, WqT, Cdim, Cdim);
  wT_f16<<<(Cdim * Cdim + 255) / 256, 256, 0, stream>>>(Wk, WkT, Cdim, Cdim);
  wT_f16<<<(Cdim * Cdim + 255) / 256, 256, 0, stream>>>(Wv, WvT, Cdim, Cdim);
  wT_f16<<<(Cdim * Cdim + 255) / 256, 256, 0, stream>>>(Wo, WoT, Cdim, Cdim);
  wT_f16<<<(Cdim * FF + 255) / 256, 256, 0, stream>>>(W1, W1T, Cdim, FF);
  wT_f16<<<(FF * Cdim + 255) / 256, 256, 0, stream>>>(W2, W2T, FF, Cdim);
  vpad_zero<<<(BATCH * HEADS * DH * 16) / 256, 256, 0, stream>>>(vb);

  const dim3 blk(128);
  // 3) Q, K, V projections
  gemm16<0><<<dim3(Mtot / 32, Cdim / 256), blk, 0, stream>>>(
      xn, WqT, bq, Mtot, Cdim, Cdim, qb, nullptr, nullptr);
  gemm16<0><<<dim3(Mtot / 32, Cdim / 256), blk, 0, stream>>>(
      xn, WkT, bk, Mtot, Cdim, Cdim, kb, nullptr, nullptr);
  gemm16<1><<<dim3(Mtot / 32, Cdim / 256), blk, 0, stream>>>(
      xn, WvT, bv, Mtot, Cdim, Cdim, vb, nullptr, nullptr);

  // 4) attention
  attn_kernel<<<dim3(Ndim / 16, BATCH * HEADS), 32, 0, stream>>>(
      qb, kb, vb, relb, relix, ao);

  // 5) output projection + residual (reads xc straight from NCHW x)
  gemm16<2><<<dim3(Mtot / 32, Cdim / 256), blk, 0, stream>>>(
      ao, WoT, bo, Mtot, Cdim, Cdim, x1h, x1, x);

  // 6) FFN
  gemm16<3><<<dim3(Mtot / 32, FF / 256), blk, 0, stream>>>(
      x1h, W1T, b1, Mtot, FF, Cdim, hid, nullptr, nullptr);
  gemm16<4><<<dim3(Mtot / 32, Cdim / 256), blk, 0, stream>>>(
      hid, W2T, b2, Mtot, Cdim, FF, nullptr, out, x1);
}
